// NeuralSplineCoupling_6047313953038
// MI455X (gfx1250) — compile-verified, matrix operand order checked by reading, not run on hardware
//
#include <hip/hip_runtime.h>
#include <hip/hip_bf16.h>

#define KNOTSN 16
#define BOUNDF 5.0f
#define BATCH 500000
#define XDIM 6
#define CDIM 4
#define HID 128
#define SPLD 47        // 3*KNOTS-1
#define LOWERD 3
#define OUTD 141       // LOWERD*SPLD
#define OUTP 144       // padded to 9 N-tiles of 16
#define TILE_M 128

typedef _Float16 v16h __attribute__((ext_vector_type(16)));
typedef _Float16 v8h  __attribute__((ext_vector_type(8)));
typedef float    v8f  __attribute__((ext_vector_type(8)));
typedef int      v4i  __attribute__((ext_vector_type(4)));

#if __has_builtin(__builtin_amdgcn_global_load_async_to_lds_b128) && \
    __has_builtin(__builtin_amdgcn_s_wait_asynccnt)
#define HAVE_ASYNC_LDS 1
typedef __attribute__((address_space(1))) v4i* gv4i_p;   // global
typedef __attribute__((address_space(3))) v4i* lv4i_p;   // LDS
#endif

// ---------------------------------------------------------------------------
// Pack f32 weights -> f16 column-major (zero padded) for WMMA B-fragments.
// w1t: [128 cols][32 k]   (k=7..31 zero)
// w2t: [128 cols][128 k]
// w3t: [144 cols][128 k]  (cols 141..143 zero)
// ---------------------------------------------------------------------------
__global__ void pack_weights_kernel(const float* __restrict__ W1,
                                    const float* __restrict__ W2,
                                    const float* __restrict__ W3,
                                    _Float16* __restrict__ w1t,
                                    _Float16* __restrict__ w2t,
                                    _Float16* __restrict__ w3t) {
    int i = blockIdx.x * 256 + threadIdx.x;
    if (i < 128 * 32) {
        int col = i >> 5, k = i & 31;
        w1t[i] = (k < 7) ? (_Float16)W1[k * HID + col] : (_Float16)0.f;
    }
    if (i < 128 * 128) {
        int col = i >> 7, k = i & 127;
        w2t[i] = (_Float16)W2[k * HID + col];
    }
    if (i < OUTP * 128) {
        int col = i >> 7, k = i & 127;
        w3t[i] = (col < OUTD) ? (_Float16)W3[k * OUTD + col] : (_Float16)0.f;
    }
}

// Load a 16-half fragment: two contiguous 8-half (16B) chunks 32B apart.
__device__ __forceinline__ v16h load_frag(const _Float16* p) {
    v8h lo = *(const v8h*)p;
    v8h hi = *(const v8h*)(p + 16);
    return __builtin_shufflevector(lo, hi, 0,1,2,3,4,5,6,7,8,9,10,11,12,13,14,15);
}

__device__ __forceinline__ v8f wmma_step(v16h a, const _Float16* bp, v8f acc) {
    v16h b = load_frag(bp);
    return __builtin_amdgcn_wmma_f32_16x16x32_f16(false, a, false, b,
                                                  (short)0, acc, false, false);
}

__device__ __forceinline__ float softplusf(float z) {
    return (z > 20.f) ? z : log1pf(expf(z));
}

// ---------------------------------------------------------------------------
// Fused MLP (3 WMMA layers) + rational-quadratic spline per 128-row tile.
// ---------------------------------------------------------------------------
__global__ __launch_bounds__(256)
void nsc_fused_kernel(const float* __restrict__ x, const float* __restrict__ c,
                      const float* __restrict__ b1, const float* __restrict__ b2,
                      const float* __restrict__ b3,
                      const _Float16* __restrict__ w1t,
                      const _Float16* __restrict__ w2t,
                      const _Float16* __restrict__ w3t,
                      float* __restrict__ out) {
    // h0/h1 dead once pbuf is written; w1 panel dead once h2 is written.
    union SharedU {
        struct { _Float16 h0[TILE_M * 32]; _Float16 h1[TILE_M * HID]; } a;
        float pbuf[TILE_M * OUTP];
    };
    union SharedU2 {
        _Float16 w1l[128 * 32];
        _Float16 h2[TILE_M * HID];
    };
    __shared__ __align__(16) SharedU  su;
    __shared__ __align__(16) SharedU2 su2;
    __shared__ __align__(16) float xtile[TILE_M * XDIM];
    __shared__ __align__(16) float ctile[TILE_M * CDIM];
    __shared__ float ldpart[TILE_M * LOWERD];

    const int tid  = threadIdx.x;
    const int lane = tid & 31;
    const int wave = tid >> 5;
    const int m0   = blockIdx.x * TILE_M;

    const int khalf = (lane >> 4) << 3;   // 0 or 8 (A/B K-chunk base)
    const int ncol  = lane & 15;          // N (or M for A-row) within tile
    const int rbase = (lane >> 4) << 3;   // C/D row sub-offset

    // ---- Stage -1: stage x/c tiles and layer-1 weight panel into LDS -------
#ifdef HAVE_ASYNC_LDS
    {
        char*       gx  = (char*)x + (size_t)m0 * (XDIM * 4);
        char*       gc  = (char*)c + (size_t)m0 * (CDIM * 4);
        const long  axb = (long)(BATCH - m0) * (XDIM * 4); // bytes valid in x tile
        const long  acb = (long)(BATCH - m0) * (CDIM * 4); // bytes valid in c tile
        if (tid < 192 && (long)tid * 16 < axb)             // 128*6*4 = 3072 B
            __builtin_amdgcn_global_load_async_to_lds_b128(
                (gv4i_p)(gx + tid * 16), (lv4i_p)((char*)xtile + tid * 16), 0, 0);
        if (tid < 128 && (long)tid * 16 < acb)             // 128*4*4 = 2048 B
            __builtin_amdgcn_global_load_async_to_lds_b128(
                (gv4i_p)(gc + tid * 16), (lv4i_p)((char*)ctile + tid * 16), 0, 0);
        // w1 panel: 128*32 halves = 8192 B = 512 chunks, 2 per thread
        char* gw = (char*)w1t;
        char* lw = (char*)su2.w1l;
        __builtin_amdgcn_global_load_async_to_lds_b128(
            (gv4i_p)(gw + tid * 16), (lv4i_p)(lw + tid * 16), 0, 0);
        __builtin_amdgcn_global_load_async_to_lds_b128(
            (gv4i_p)(gw + 4096 + tid * 16), (lv4i_p)(lw + 4096 + tid * 16), 0, 0);
        __builtin_amdgcn_s_wait_asynccnt(0);
    }
#else
    for (int i = tid; i < TILE_M * XDIM; i += 256) {
        int g = m0 * XDIM + i;
        xtile[i] = (g < BATCH * XDIM) ? x[g] : 0.f;
    }
    for (int i = tid; i < TILE_M * CDIM; i += 256) {
        int g = m0 * CDIM + i;
        ctile[i] = (g < BATCH * CDIM) ? c[g] : 0.f;
    }
    for (int i = tid; i < 128 * 32; i += 256) su2.w1l[i] = w1t[i];
#endif
    __syncthreads();

    // ---- Stage 0: h0 = [upper(3) | c(4) | zeros] as f16, K padded to 32 ----
    if (tid < TILE_M) {
        int  row = m0 + tid;
        bool ok  = row < BATCH;
        _Float16* hr = &su.a.h0[tid * 32];
        #pragma unroll
        for (int j = 0; j < LOWERD; ++j)
            hr[j] = ok ? (_Float16)xtile[tid * XDIM + LOWERD + j] : (_Float16)0.f;
        #pragma unroll
        for (int j = 0; j < CDIM; ++j)
            hr[LOWERD + j] = ok ? (_Float16)ctile[tid * CDIM + j] : (_Float16)0.f;
        #pragma unroll
        for (int j = LOWERD + CDIM; j < 32; ++j) hr[j] = (_Float16)0.f;
    }
    __syncthreads();

    // ---- Stage 1: h1 = relu(h0 @ W1 + b1)   (K=32 single WMMA step) --------
    {
        const int arow = wave * 16 + ncol;
        v16h a = load_frag(&su.a.h0[arow * 32 + khalf]);
        #pragma unroll
        for (int nt = 0; nt < 8; ++nt) {
            int col = nt * 16 + ncol;
            v8f acc = {};
            acc = wmma_step(a, &su2.w1l[col * 32 + khalf], acc);
            float bias = b1[col];
            #pragma unroll
            for (int r = 0; r < 8; ++r) {
                float v = acc[r] + bias;
                v = v > 0.f ? v : 0.f;
                su.a.h1[(wave * 16 + rbase + r) * HID + col] = (_Float16)v;
            }
        }
    }
    __syncthreads();

    // ---- Stage 2: h2 = relu(h1 @ W2 + b2)   (K=128 -> 4 WMMA steps) --------
    {
        const int arow = wave * 16 + ncol;
        v16h a0 = load_frag(&su.a.h1[arow * HID +  0 + khalf]);
        v16h a1 = load_frag(&su.a.h1[arow * HID + 32 + khalf]);
        v16h a2 = load_frag(&su.a.h1[arow * HID + 64 + khalf]);
        v16h a3 = load_frag(&su.a.h1[arow * HID + 96 + khalf]);
        #pragma unroll
        for (int nt = 0; nt < 8; ++nt) {
            int col = nt * 16 + ncol;
            const _Float16* bp = &w2t[col * HID + khalf];
            v8f acc = {};
            acc = wmma_step(a0, bp +  0, acc);
            acc = wmma_step(a1, bp + 32, acc);
            acc = wmma_step(a2, bp + 64, acc);
            acc = wmma_step(a3, bp + 96, acc);
            float bias = b2[col];
            #pragma unroll
            for (int r = 0; r < 8; ++r) {
                float v = acc[r] + bias;
                v = v > 0.f ? v : 0.f;
                su2.h2[(wave * 16 + rbase + r) * HID + col] = (_Float16)v;
            }
        }
    }
    __syncthreads();

    // ---- Stage 3: p = h2 @ W3 + b3  (N padded to 144, f32 into pbuf) -------
    {
        const int arow = wave * 16 + ncol;
        v16h a0 = load_frag(&su2.h2[arow * HID +  0 + khalf]);
        v16h a1 = load_frag(&su2.h2[arow * HID + 32 + khalf]);
        v16h a2 = load_frag(&su2.h2[arow * HID + 64 + khalf]);
        v16h a3 = load_frag(&su2.h2[arow * HID + 96 + khalf]);
        #pragma unroll
        for (int nt = 0; nt < 9; ++nt) {
            int col = nt * 16 + ncol;
            const _Float16* bp = &w3t[col * HID + khalf];
            v8f acc = {};
            acc = wmma_step(a0, bp +  0, acc);
            acc = wmma_step(a1, bp + 32, acc);
            acc = wmma_step(a2, bp + 64, acc);
            acc = wmma_step(a3, bp + 96, acc);
            float bias = (col < OUTD) ? b3[col] : 0.f;
            #pragma unroll
            for (int r = 0; r < 8; ++r)
                su.pbuf[(wave * 16 + rbase + r) * OUTP + col] = acc[r] + bias;
        }
    }
    __syncthreads();

    // ---- Stage 4: rational-quadratic spline per (row, dim) -----------------
    for (int t = tid; t < TILE_M * LOWERD; t += 256) {
        int r = t / LOWERD;
        int d = t - r * LOWERD;
        int row = m0 + r;
        if (row < BATCH) {
            const float* pw = &su.pbuf[r * OUTP + d * SPLD];
            const float* ph = pw + KNOTSN;
            const float* pd = pw + 2 * KNOTSN;

            float mw = pw[0], mh = ph[0];
            #pragma unroll
            for (int k = 1; k < KNOTSN; ++k) {
                mw = fmaxf(mw, pw[k]);
                mh = fmaxf(mh, ph[k]);
            }
            float sw = 0.f, sh = 0.f;
            #pragma unroll
            for (int k = 0; k < KNOTSN; ++k) {
                sw += expf(pw[k] - mw);
                sh += expf(ph[k] - mh);
            }
            float tw = (2.f * BOUNDF) / sw;
            float th = (2.f * BOUNDF) / sh;

            float xin = xtile[r * XDIM + d];
            bool  oob = (xin <= -BOUNDF) || (xin >= BOUNDF);
            float xm  = oob ? -BOUNDF : xin;

            // streaming cumsum + bin select (idx = last k with knot_k <= xm)
            float xc = -BOUNDF, yc = -BOUNDF;
            float xl = -BOUNDF, yl = -BOUNDF, wk = 1.f, hk = 1.f;
            int idx = 0;
            #pragma unroll
            for (int k = 0; k < KNOTSN; ++k) {
                float wkk = expf(pw[k] - mw) * tw;
                float hkk = expf(ph[k] - mh) * th;
                if (xm >= xc) { idx = k; xl = xc; yl = yc; wk = wkk; hk = hkk; }
                xc += wkk; yc += hkk;
            }
            float dl = (idx == 0)          ? 1.f : softplusf(pd[idx - 1]);
            float dr = (idx == KNOTSN - 1) ? 1.f : softplusf(pd[idx]);

            float sk   = hk / wk;
            float relx = fminf(fmaxf((xm - xl) / wk, 0.f), 1.f);
            float r1   = relx * (1.f - relx);
            float den  = sk + (dr + dl - 2.f * sk) * r1;
            float num  = hk * (sk * relx * relx + dl * r1);
            float y    = yl + num / den;
            float omr  = 1.f - relx;
            float ld   = 2.f * logf(sk)
                       + logf(dr * relx * relx + 2.f * sk * r1 + dl * omr * omr)
                       - 2.f * logf(den);
            if (oob) { y = xin; ld = 0.f; }

            out[row * XDIM + d] = y;
            ldpart[t] = ld;
        } else {
            ldpart[t] = 0.f;
        }
    }
    __syncthreads();

    // ---- Stage 5: log-det reduction + pass-through of upper dims -----------
    if (tid < TILE_M) {
        int row = m0 + tid;
        if (row < BATCH) {
            out[BATCH * XDIM + row] = ldpart[tid * LOWERD + 0]
                                    + ldpart[tid * LOWERD + 1]
                                    + ldpart[tid * LOWERD + 2];
            #pragma unroll
            for (int j = 0; j < LOWERD; ++j)
                out[row * XDIM + LOWERD + j] = xtile[tid * XDIM + LOWERD + j];
        }
    }
}

extern "C" void kernel_launch(void* const* d_in, const int* in_sizes, int n_in,
                              void* d_out, int out_size, void* d_ws, size_t ws_size,
                              hipStream_t stream) {
    const float* x  = (const float*)d_in[0];
    const float* c  = (const float*)d_in[1];
    const float* W1 = (const float*)d_in[2];
    const float* b1 = (const float*)d_in[3];
    const float* W2 = (const float*)d_in[4];
    const float* b2 = (const float*)d_in[5];
    const float* W3 = (const float*)d_in[6];
    const float* b3 = (const float*)d_in[7];
    float* out = (float*)d_out;

    _Float16* w1t = (_Float16*)d_ws;          // 128*32  halves
    _Float16* w2t = w1t + 128 * 32;           // 128*128 halves
    _Float16* w3t = w2t + 128 * 128;          // 144*128 halves

    pack_weights_kernel<<<(OUTP * 128 + 255) / 256, 256, 0, stream>>>(
        W1, W2, W3, w1t, w2t, w3t);

    int grid = (BATCH + TILE_M - 1) / TILE_M;
    nsc_fused_kernel<<<grid, 256, 0, stream>>>(x, c, b1, b2, b3, w1t, w2t, w3t, out);
}
